// PrimalModel_82575041232931
// MI455X (gfx1250) — matrix operand order repaired and checked by reading, not compiled, not run on hardware
//
#include <hip/hip_runtime.h>

#define G 1024
#define NN 40
#define M_PAIRS 780
#define TILES_PER_G 49   // ceil(780/16)

typedef __attribute__((ext_vector_type(16))) _Float16 v16h;
typedef __attribute__((ext_vector_type(8)))  _Float16 v8h;
typedef __attribute__((ext_vector_type(8)))  float    v8f;

union V16U { v16h v; v8h h[2]; };

__device__ __forceinline__ float lrelu(float v) { return v > 0.f ? v : 0.1f * v; }

__device__ __forceinline__ v8f zero8() { v8f z; for (int i = 0; i < 8; ++i) z[i] = 0.f; return z; }

__device__ __forceinline__ v8f wmma_f16(v16h a, v16h b, v8f c) {
  // D = A(16x32 f16) x B(32x16 f16) + C(16x16 f32)
  return __builtin_amdgcn_wmma_f32_16x16x32_f16(false, a, false, b, (short)0, c, false, false);
}

// A-matrix fragment (16x32 f16) from row-major LDS tile [16][64].
// lanes 0-15 hold M=lane, K = {kb+0..7, kb+16..23}; lanes 16-31: K = {kb+8..15, kb+24..31}
__device__ __forceinline__ v16h a_frag_lds(const _Float16* A, int lane, int kb) {
  const int m = lane & 15, hi = lane >> 4;
  V16U r;
  r.h[0] = *(const v8h*)(A + m * 64 + kb + hi * 8);
  r.h[1] = *(const v8h*)(A + m * 64 + kb + 16 + hi * 8);
  return r.v;
}

// B-matrix fragment (32x16 f16) from TRANSPOSED weights in LDS: WT[n*64 + k].
// lanes 0-15: N=lane, K={kb+0..7, kb+16..23}; lanes 16-31: N=lane-16, K={kb+8..15, kb+24..31}
__device__ __forceinline__ v16h b_frag_lds(const _Float16* WT, int lane, int ntile, int kb) {
  const int n = ntile * 16 + (lane & 15), hi = lane >> 4;
  const _Float16* p = WT + n * 64 + kb + hi * 8;
  V16U r;
  r.h[0] = *(const v8h*)(p);
  r.h[1] = *(const v8h*)(p + 16);
  return r.v;
}

// ---------------- Kernel A: GCN (analytic norm = 1/39 on complete graph) ----------------
__global__ __launch_bounds__(256) void gcn_kernel(
    const float* __restrict__ x,
    const float* __restrict__ Wg0, const float* __restrict__ bg0,
    const float* __restrict__ Wg1, const float* __restrict__ bg1,
    const float* __restrict__ Wg2, const float* __restrict__ bg2,
    float* __restrict__ h_out)
{
  __shared__ float sA[NN * 32];
  __shared__ float sB[NN * 32];
  __shared__ float sm[NN * 64];
  __shared__ float sW[32 * 64];
  __shared__ float sb[64];
  __shared__ float sS[64];
  const int tid = threadIdx.x;
  const int g = blockIdx.x;
  const float invdeg = 1.0f / 39.0f;

  for (int idx = tid; idx < NN * 6; idx += 256) {
    int n = idx / 6, c = idx - n * 6;
    sA[n * 6 + c] = x[(g * NN + n) * 16 + c];
  }
  __syncthreads();

  auto layer = [&](const float* W, const float* b, int Cin, int Cout,
                   const float* in, float* out, bool act, bool last) {
    for (int idx = tid; idx < Cin * Cout; idx += 256) sW[idx] = W[idx];
    for (int idx = tid; idx < Cout;       idx += 256) sb[idx] = b[idx];
    __syncthreads();
    for (int idx = tid; idx < NN * Cout; idx += 256) {
      int n = idx / Cout, d = idx - n * Cout;
      float s = 0.f;
      for (int c = 0; c < Cin; ++c) s += in[n * Cin + c] * sW[c * Cout + d];
      sm[idx] = s;
    }
    __syncthreads();
    for (int d = tid; d < Cout; d += 256) {
      float s = 0.f;
      for (int n = 0; n < NN; ++n) s += sm[n * Cout + d];
      sS[d] = s;
    }
    __syncthreads();
    for (int idx = tid; idx < NN * Cout; idx += 256) {
      int n = idx / Cout, d = idx - n * Cout;
      float v = (sS[d] - sm[idx]) * invdeg + sb[d];
      if (act) v = lrelu(v);
      if (last) h_out[(size_t)(g * NN + n) * 64 + d] = v;
      else      out[idx] = v;
    }
    __syncthreads();
  };

  layer(Wg0, bg0, 6, 32, sA, sB, true, false);
  layer(Wg1, bg1, 32, 32, sB, sA, true, false);
  layer(Wg2, bg2, 32, 64, sA, nullptr, false, true);
}

// ---------------- Kernel B: node MLP (vp) + factored edge layer-1 (t = h @ We0) --------
__global__ __launch_bounds__(256) void node_kernel(
    const float* __restrict__ h,
    const float* __restrict__ Wn0, const float* __restrict__ bn0,
    const float* __restrict__ Wn1, const float* __restrict__ bn1,
    const float* __restrict__ Wn2, const float* __restrict__ bn2,
    const float* __restrict__ We0,
    float* __restrict__ t_out, float* __restrict__ vp_out)
{
  __shared__ __align__(16) _Float16 WT0[64 * 64];
  __shared__ __align__(16) _Float16 WT1[64 * 64];
  __shared__ __align__(16) _Float16 WT2[16 * 64];
  __shared__ __align__(16) _Float16 WTe[64 * 64];
  __shared__ float sb0[64], sb1[64], sb2[16];
  __shared__ __align__(16) _Float16 Ab[8][16 * 64];
  const int tid = threadIdx.x;

  for (int idx = tid; idx < 64 * 64; idx += 256) {
    int n = idx >> 6, k = idx & 63;
    WT0[idx] = (_Float16)Wn0[k * 64 + n];
    WT1[idx] = (_Float16)Wn1[k * 64 + n];
    WTe[idx] = (_Float16)We0[k * 64 + n];
  }
  for (int idx = tid; idx < 16 * 64; idx += 256) {
    int n = idx >> 6, k = idx & 63;
    WT2[idx] = (n < 10) ? (_Float16)Wn2[k * 10 + n] : (_Float16)0.f;
  }
  for (int d = tid; d < 64; d += 256) { sb0[d] = bn0[d]; sb1[d] = bn1[d]; }
  for (int d = tid; d < 16; d += 256) sb2[d] = (d < 10) ? bn2[d] : 0.f;
  __syncthreads();

  const int w = tid >> 5, lane = tid & 31;
  const int hi = lane >> 4, colb = lane & 15;
  const int base = (blockIdx.x * 8 + w) * 16;   // node-tile base (16 nodes per wave)
  _Float16* A = Ab[w];

  for (int idx = lane; idx < 16 * 64; idx += 32)
    A[idx] = (_Float16)h[(size_t)(base + (idx >> 6)) * 64 + (idx & 63)];

  // t = A @ We0 and z0 = lrelu(A @ Wn0 + bn0), sharing A fragments
  v8f accT[4], acc0[4];
  for (int n = 0; n < 4; ++n) { accT[n] = zero8(); acc0[n] = zero8(); }
  for (int kc = 0; kc < 2; ++kc) {
    v16h a = a_frag_lds(A, lane, kc * 32);
    for (int n = 0; n < 4; ++n) {
      accT[n] = wmma_f16(a, b_frag_lds(WTe, lane, n, kc * 32), accT[n]);
      acc0[n] = wmma_f16(a, b_frag_lds(WT0, lane, n, kc * 32), acc0[n]);
    }
  }
  for (int n = 0; n < 4; ++n)
    for (int r = 0; r < 8; ++r) {
      int row = r + hi * 8, col = n * 16 + colb;
      t_out[(size_t)(base + row) * 64 + col] = accT[n][r];
    }
  for (int n = 0; n < 4; ++n)
    for (int r = 0; r < 8; ++r) {
      int row = r + hi * 8, col = n * 16 + colb;
      A[row * 64 + col] = (_Float16)lrelu(acc0[n][r] + sb0[col]);
    }

  // z1 = lrelu(z0 @ Wn1 + bn1)
  v8f acc1[4];
  for (int n = 0; n < 4; ++n) acc1[n] = zero8();
  for (int kc = 0; kc < 2; ++kc) {
    v16h a = a_frag_lds(A, lane, kc * 32);
    for (int n = 0; n < 4; ++n)
      acc1[n] = wmma_f16(a, b_frag_lds(WT1, lane, n, kc * 32), acc1[n]);
  }
  for (int n = 0; n < 4; ++n)
    for (int r = 0; r < 8; ++r) {
      int row = r + hi * 8, col = n * 16 + colb;
      A[row * 64 + col] = (_Float16)lrelu(acc1[n][r] + sb1[col]);
    }

  // vp = z1 @ Wn2 + bn2 (10 cols)
  v8f acc2 = zero8();
  for (int kc = 0; kc < 2; ++kc) {
    v16h a = a_frag_lds(A, lane, kc * 32);
    acc2 = wmma_f16(a, b_frag_lds(WT2, lane, 0, kc * 32), acc2);
  }
  if (colb < 10)
    for (int r = 0; r < 8; ++r) {
      int row = r + hi * 8;
      vp_out[(size_t)(base + row) * 10 + colb] = acc2[r] + sb2[colb];
    }
}

// ------- Kernel C: per-graph edge MLP + assembly of X (off-diagonal + diagonal) --------
// One block per graph: stage the graph's t block (40x64 f32, 10KB) in LDS once,
// 8 waves sweep the 49 edge-tiles; diagonal blocks written from vp at the end.
__global__ __launch_bounds__(256) void edge_kernel(
    const int* __restrict__ ud,
    const float* __restrict__ t,
    const float* __restrict__ vp,
    const float* __restrict__ be0,
    const float* __restrict__ We1, const float* __restrict__ be1,
    const float* __restrict__ We2, const float* __restrict__ be2,
    float* __restrict__ X)
{
  __shared__ __align__(16) _Float16 WT1[64 * 64];
  __shared__ __align__(16) _Float16 WT2[16 * 64];
  __shared__ float sbe0[64], sbe1[64], sbe2[16];
  __shared__ __align__(16) float Tg[NN * 64];          // group's t block
  __shared__ __align__(16) _Float16 Zb[8][16 * 64];
  __shared__ float Eb[8][16 * 16];
  const int tid = threadIdx.x;
  const int g = blockIdx.x;

  for (int idx = tid; idx < 64 * 64; idx += 256) {
    int n = idx >> 6, k = idx & 63;
    WT1[idx] = (_Float16)We1[k * 64 + n];
  }
  for (int idx = tid; idx < 16 * 64; idx += 256) {
    int n = idx >> 6, k = idx & 63;
    WT2[idx] = (n < 10) ? (_Float16)We2[k * 10 + n] : (_Float16)0.f;
  }
  for (int d = tid; d < 64; d += 256) { sbe0[d] = be0[d]; sbe1[d] = be1[d]; }
  for (int d = tid; d < 16; d += 256) sbe2[d] = (d < 10) ? be2[d] : 0.f;
  // stage t for this graph as float4s
  {
    const float4* src = (const float4*)(t + (size_t)g * NN * 64);
    float4* dst = (float4*)Tg;
    for (int idx = tid; idx < NN * 16; idx += 256) dst[idx] = src[idx];
  }
  __syncthreads();

  const int w = tid >> 5, lane = tid & 31;
  const int hi = lane >> 4, colb = lane & 15;
  float* Xg = X + (size_t)g * 160 * 160;
  _Float16* Z = Zb[w];
  float* E = Eb[w];
  const int SM[4][4] = {{0,1,2,3},{1,4,5,6},{2,5,7,8},{3,6,8,9}};

  for (int tile = w; tile < TILES_PER_G; tile += 8) {
    const int tb = tile * 16;
    const int p  = tb + colb;
    const int pc = p < M_PAIRS ? p : M_PAIRS - 1;
    const int ei = ud[2 * pc], ej = ud[2 * pc + 1];
    const float* ti = Tg + ei * 64;
    const float* tj = Tg + ej * 64;

    // z1 = lrelu(t_i + t_j + be0), built directly as A fragments (LDS b128 runs)
    v16h afr[2];
    for (int kc = 0; kc < 2; ++kc) {
      V16U r;
      for (int run = 0; run < 2; ++run) {
        const int k0 = kc * 32 + run * 16 + hi * 8;
        float u[8], vvv[8];
        *(float4*)&u[0]   = *(const float4*)(ti + k0);
        *(float4*)&u[4]   = *(const float4*)(ti + k0 + 4);
        *(float4*)&vvv[0] = *(const float4*)(tj + k0);
        *(float4*)&vvv[4] = *(const float4*)(tj + k0 + 4);
        v8h hh;
        for (int e = 0; e < 8; ++e)
          hh[e] = (_Float16)lrelu(u[e] + vvv[e] + sbe0[k0 + e]);
        r.h[run] = hh;
      }
      afr[kc] = r.v;
    }

    // z2 = lrelu(z1 @ We1 + be1)
    v8f acc[4];
    for (int n = 0; n < 4; ++n) acc[n] = zero8();
    for (int kc = 0; kc < 2; ++kc)
      for (int n = 0; n < 4; ++n)
        acc[n] = wmma_f16(afr[kc], b_frag_lds(WT1, lane, n, kc * 32), acc[n]);
    for (int n = 0; n < 4; ++n)
      for (int r = 0; r < 8; ++r) {
        int row = r + hi * 8, col = n * 16 + colb;
        Z[row * 64 + col] = (_Float16)lrelu(acc[n][r] + sbe1[col]);
      }

    // ep = z2 @ We2 + be2 (10 cols) -> stage per-edge rows in LDS
    v8f acc2 = zero8();
    for (int kc = 0; kc < 2; ++kc) {
      v16h a = a_frag_lds(Z, lane, kc * 32);
      acc2 = wmma_f16(a, b_frag_lds(WT2, lane, 0, kc * 32), acc2);
    }
    for (int r = 0; r < 8; ++r)
      E[(r + hi * 8) * 16 + colb] = acc2[r] + sbe2[colb];

    // Scatter 4x4 symmetric blocks: lanes 0-15 -> block (i,j), lanes 16-31 -> (j,i)
    const int pe = tb + colb;
    if (pe < M_PAIRS) {
      int ii = ud[2 * pe], jj = ud[2 * pe + 1];
      if (hi) { int tmp = ii; ii = jj; jj = tmp; }
      const float* ev = E + colb * 16;
      for (int a = 0; a < 4; ++a) {
        float4 vrow = make_float4(ev[SM[a][0]], ev[SM[a][1]], ev[SM[a][2]], ev[SM[a][3]]);
        *(float4*)(Xg + (size_t)(4 * ii + a) * 160 + 4 * jj) = vrow;
      }
    }
  }

  // Diagonal 4x4 blocks from vp (one thread per node)
  for (int n = tid; n < NN; n += 256) {
    float ev[10];
    for (int c = 0; c < 10; ++c) ev[c] = vp[(size_t)(g * NN + n) * 10 + c];
    for (int a = 0; a < 4; ++a) {
      float4 vrow = make_float4(ev[SM[a][0]], ev[SM[a][1]], ev[SM[a][2]], ev[SM[a][3]]);
      *(float4*)(Xg + (size_t)(4 * n + a) * 160 + 4 * n) = vrow;
    }
  }
}

extern "C" void kernel_launch(void* const* d_in, const int* in_sizes, int n_in,
                              void* d_out, int out_size, void* d_ws, size_t ws_size,
                              hipStream_t stream) {
  (void)in_sizes; (void)n_in; (void)out_size; (void)ws_size;
  const float* x   = (const float*)d_in[0];
  const int*   ud  = (const int*)d_in[2];          // ud_edges (780,2)
  const float* Wg0 = (const float*)d_in[4];  const float* bg0 = (const float*)d_in[5];
  const float* Wg1 = (const float*)d_in[6];  const float* bg1 = (const float*)d_in[7];
  const float* Wg2 = (const float*)d_in[8];  const float* bg2 = (const float*)d_in[9];
  const float* Wn0 = (const float*)d_in[10]; const float* bn0 = (const float*)d_in[11];
  const float* Wn1 = (const float*)d_in[12]; const float* bn1 = (const float*)d_in[13];
  const float* Wn2 = (const float*)d_in[14]; const float* bn2 = (const float*)d_in[15];
  const float* We0 = (const float*)d_in[16]; const float* be0 = (const float*)d_in[17];
  const float* We1 = (const float*)d_in[18]; const float* be1 = (const float*)d_in[19];
  const float* We2 = (const float*)d_in[20]; const float* be2 = (const float*)d_in[21];

  float* h_out = (float*)d_out;                    // (G*N, 64)
  float* X     = h_out + (size_t)G * NN * 64;      // (G, 160, 160)
  float* t     = (float*)d_ws;                     // (G*N, 64) factored edge layer-1
  float* vp    = t + (size_t)G * NN * 64;          // (G*N, 10)

  gcn_kernel <<<G, 256, 0, stream>>>(x, Wg0, bg0, Wg1, bg1, Wg2, bg2, h_out);
  node_kernel<<<(G * NN / 16) / 8, 256, 0, stream>>>(h_out, Wn0, bn0, Wn1, bn1,
                                                     Wn2, bn2, We0, t, vp);
  edge_kernel<<<G, 256, 0, stream>>>(ud, t, vp, be0, We1, be1, We2, be2, X);
}
